// Attention_20813411516578
// MI455X (gfx1250) — matrix operand order repaired and checked by reading, not compiled, not run on hardware
//
#include <hip/hip_runtime.h>

// ---- problem constants (match reference) ----
#define NDIM   768
#define QKVOUT 2304
#define NSEQ   2048
#define NBATCH 4
#define NHEADS 12
#define HDIM   64
#define QELE   ((size_t)NBATCH * NHEADS * NSEQ * HDIM)   // 6,291,456 elems per q/k/v
#define XELE   ((size_t)NBATCH * NSEQ * NDIM)            // 6,291,456
#define QWELE  ((size_t)QKVOUT * NDIM)                   // 1,769,472
#define PWELE  ((size_t)NDIM * NDIM)                     // 589,824

typedef __attribute__((ext_vector_type(16))) __bf16 v16bf;
typedef __attribute__((ext_vector_type(8)))  __bf16 v8bf;
typedef __attribute__((ext_vector_type(8)))  float  v8f;
typedef __attribute__((ext_vector_type(4)))  float  f32x4;
typedef __attribute__((ext_vector_type(4)))  unsigned u32x4;
typedef __attribute__((ext_vector_type(8)))  int    i32x8;
typedef __attribute__((ext_vector_type(4)))  int    i32x4;

union U16 { v16bf v; v8bf h[2]; };

// Native f32->bf16 (lowers to hardware cvt on gfx1250, RTNE)
__device__ inline __bf16 f2bf(float f) { return (__bf16)f; }

__device__ inline v8f wmma_bf16(v16bf a, v16bf b, v8f c) {
  return __builtin_amdgcn_wmma_f32_16x16x32_bf16(
      /*neg_a=*/false, a, /*neg_b=*/false, b,
      /*c_mod=*/(short)0, c, /*reuse_a=*/false, /*reuse_b=*/false);
}

__device__ inline float rmax16(float v) {
#pragma unroll
  for (int m = 1; m < 16; m <<= 1) v = fmaxf(v, __shfl_xor(v, m, 32));
  return v;
}
__device__ inline float rsum16(float v) {
#pragma unroll
  for (int m = 1; m < 16; m <<= 1) v += __shfl_xor(v, m, 32);
  return v;
}

// ---------------------------------------------------------------------------
// Kernel 0: one-shot fp32 -> bf16 convert (8 elems/thread, b128 in, b128 out)
// ---------------------------------------------------------------------------
__global__ __launch_bounds__(256) void cvt_bf16_kernel(
    const float* __restrict__ src, __bf16* __restrict__ dst, int n8) {
  int i = blockIdx.x * 256 + threadIdx.x;
  if (i >= n8) return;
  f32x4 a = *(const f32x4*)(src + (size_t)i * 8);
  f32x4 b = *(const f32x4*)(src + (size_t)i * 8 + 4);
  v8bf r;
#pragma unroll
  for (int j = 0; j < 4; ++j) { r[j] = (__bf16)a[j]; r[4 + j] = (__bf16)b[j]; }
  *(v8bf*)(dst + (size_t)i * 8) = r;
}

// ---------------------------------------------------------------------------
// Kernel 1: y = x @ qkv_w.T + qkv_b  (all-bf16 inputs) -> [3][B*H][N][64] bf16
// Wave computes a 16x64 output strip; k-loop step 32.
// ---------------------------------------------------------------------------
__global__ __launch_bounds__(256) void qkv_gemm_kernel(
    const __bf16* __restrict__ x, const __bf16* __restrict__ w,
    const float* __restrict__ bias, __bf16* __restrict__ qkv_ws) {
  const int wv   = threadIdx.x >> 5;
  const int lane = threadIdx.x & 31;
  const int ncol = lane & 15;
  const int hi   = lane >> 4;

  const int m0 = (blockIdx.x % 64) * 128 + wv * 16;   // 8192 rows / 16
  const int o0 = (blockIdx.x / 64) * 64;              // 2304 cols / 64

  const int m = m0 + ncol;                            // A-frag row of this lane
  const __bf16* xrow = x + (size_t)m * NDIM;

  v8f acc[4] = {};

  for (int k0 = 0; k0 < NDIM; k0 += 32) {
    U16 au;   // A: elem e <- K=hi*16+e
    au.h[0] = *(const v8bf*)(xrow + k0 + hi * 16);
    au.h[1] = *(const v8bf*)(xrow + k0 + hi * 16 + 8);
#pragma unroll
    for (int nt = 0; nt < 4; ++nt) {
      // B: elem e <- W[o][k0 + hi*16 + e]  (16 contiguous bf16 = 32B)
      const __bf16* wrow = w + (size_t)(o0 + nt * 16 + ncol) * NDIM + k0 + hi * 16;
      U16 bu;
      bu.h[0] = *(const v8bf*)(wrow);
      bu.h[1] = *(const v8bf*)(wrow + 8);
      acc[nt] = wmma_bf16(au.v, bu.v, acc[nt]);
    }
  }

  // Scatter store (bias added) into q/k/v sections laid out [B*H][N][64] bf16.
#pragma unroll
  for (int nt = 0; nt < 4; ++nt) {
    const int o = o0 + nt * 16 + ncol;
    const float bv = bias[o];
    const int which = o / NDIM;          // 0=q 1=k 2=v
    const int c     = o - which * NDIM;
    const int h     = c >> 6;
    const int d     = c & 63;
    __bf16* base = qkv_ws + (size_t)which * QELE;
#pragma unroll
    for (int r = 0; r < 8; ++r) {
      const int row = m0 + hi * 8 + r;
      const int bb  = row >> 11;         // /2048
      const int n   = row & 2047;
      base[((size_t)(bb * NHEADS + h) * NSEQ + n) * HDIM + d] =
          f2bf(acc[nt][r] + bv);
    }
  }
}

// ---------------------------------------------------------------------------
// Kernel 2: flash attention per (b,h). Block = 8 waves = 128 q rows.
// K tile staged by TDM tensor_load_to_lds when available (else cooperative);
// V staged transposed for contiguous PV B-frags. Online softmax.
// ---------------------------------------------------------------------------
__global__ __launch_bounds__(256) void attn_kernel(
    const __bf16* __restrict__ qkv_ws, __bf16* __restrict__ attn_ws) {
  __shared__ __bf16 Klds[32 * 64];      // [jc][d]
  __shared__ __bf16 Vtlds[64 * 32];     // [d][jc]  (transposed V tile)
  __shared__ __bf16 Plds[8 * 16 * 32];  // per-wave 16x32 P tile

  const int tid  = threadIdx.x;
  const int wv   = tid >> 5;
  const int lane = tid & 31;
  const int ncol = lane & 15;
  const int hi   = lane >> 4;

  const int qt = blockIdx.x & 15;       // 16 q-tiles of 128 rows
  const int bh = blockIdx.x >> 4;       // b*NHEADS + h
  const int bb = bh / NHEADS;
  const int hh = bh - bb * NHEADS;

  const __bf16* Q = qkv_ws + (size_t)bh * NSEQ * HDIM;
  const __bf16* K = qkv_ws + QELE + (size_t)bh * NSEQ * HDIM;
  const __bf16* V = qkv_ws + 2 * QELE + (size_t)bh * NSEQ * HDIM;

  const int q0   = qt * 128 + wv * 16;
  const int mrow = q0 + ncol;

  // Q fragments (held in registers for the whole j loop)
  v16bf qa[2];
#pragma unroll
  for (int kk = 0; kk < 2; ++kk) {
    U16 u;
    const __bf16* qb = Q + (size_t)mrow * HDIM + kk * 32;
    u.h[0] = *(const v8bf*)(qb + hi * 16);
    u.h[1] = *(const v8bf*)(qb + hi * 16 + 8);
    qa[kk] = u.v;
  }

  float mprev[8], lsum[8];
#pragma unroll
  for (int r = 0; r < 8; ++r) { mprev[r] = -1e30f; lsum[r] = 0.0f; }
  v8f acc[4] = {};

  const float scale = 0.125f;           // 1/sqrt(64)
  const int row = tid >> 3;             // 0..31 (cooperative loader row)
  const int c8  = tid & 7;              // 0..7

#if __has_builtin(__builtin_amdgcn_tensor_load_to_lds)
  const unsigned klds_off = (unsigned)(unsigned long long)(const void*)&Klds[0];
#endif

  for (int j = 0; j < NSEQ; j += 32) {
    // ---- K tile (32x64 bf16) into LDS ----
#if __has_builtin(__builtin_amdgcn_tensor_load_to_lds)
    if (wv == 0) {
      unsigned long long ga = (unsigned long long)(const void*)(K + (size_t)j * HDIM);
      u32x4 g0;
      g0[0] = 1u;                                   // count=1, is_restore=0
      g0[1] = klds_off;                             // lds_addr
      g0[2] = (unsigned)(ga & 0xffffffffu);         // global_addr[31:0]
      g0[3] = (unsigned)((ga >> 32) & 0x1ffffffu) | (2u << 30);  // addr hi | type=2
      i32x8 g1;
      g1[0] = (1 << 16);                            // data_size=2B, mask=0
      g1[1] = (HDIM << 16);                         // tensor_dim0=64 (lo16 @ bits 63:48)
      g1[2] = (NSEQ << 16);                         // tensor_dim0 hi=0 | tensor_dim1 lo=2048
      g1[3] = (HDIM << 16);                         // tensor_dim1 hi=0 | tile_dim0=64
      g1[4] = 32;                                   // tile_dim1=32, tile_dim2=0
      g1[5] = HDIM;                                 // tensor_dim0_stride=64 (bits 207:160)
      g1[6] = 0;                                    // stride hi | tensor_dim1_stride lo
      g1[7] = 0;
      i32x4 gz = {0, 0, 0, 0};
      i32x8 gz8 = {0, 0, 0, 0, 0, 0, 0, 0};
      __builtin_amdgcn_tensor_load_to_lds(g0, g1, gz, gz, gz8, 0);
      __builtin_amdgcn_s_wait_tensorcnt(0);
    }
#else
    {
      v8bf kvec = *(const v8bf*)(K + (size_t)(j + row) * HDIM + c8 * 8);
      *(v8bf*)(Klds + row * 64 + c8 * 8) = kvec;
    }
#endif
    // ---- V tile, transposed into LDS ----
    v8bf vvec = *(const v8bf*)(V + (size_t)(j + row) * HDIM + c8 * 8);
#pragma unroll
    for (int i = 0; i < 8; ++i) Vtlds[(c8 * 8 + i) * 32 + row] = vvec[i];
    if (j + 32 < NSEQ) {                 // prefetch next tiles
      __builtin_prefetch(K + (size_t)(j + 32 + row) * HDIM + c8 * 8, 0, 1);
      __builtin_prefetch(V + (size_t)(j + 32 + row) * HDIM + c8 * 8, 0, 1);
    }
    __syncthreads();

    // ---- S = Q K^T (16 x 32), two 16x16 C tiles, k-dim = 64 ----
    v8f s[2];
#pragma unroll
    for (int nt = 0; nt < 2; ++nt) {
      v8f c = {};
#pragma unroll
      for (int kk = 0; kk < 2; ++kk) {
        U16 u;
        const __bf16* kb = Klds + (nt * 16 + ncol) * 64 + kk * 32 + hi * 16;
        u.h[0] = *(const v8bf*)(kb);
        u.h[1] = *(const v8bf*)(kb + 8);
        c = wmma_bf16(qa[kk], u.v, c);
      }
      s[nt] = c * scale;
    }

    // ---- online softmax (per-row stats across 16 lanes of each half) ----
    float alpha[8];
#pragma unroll
    for (int r = 0; r < 8; ++r) {
      float rv = rmax16(fmaxf(s[0][r], s[1][r]));
      float mnew = fmaxf(mprev[r], rv);
      alpha[r] = __expf(mprev[r] - mnew);
      mprev[r] = mnew;
      float p0 = __expf(s[0][r] - mnew);
      float p1 = __expf(s[1][r] - mnew);
      s[0][r] = p0; s[1][r] = p1;
      lsum[r] = lsum[r] * alpha[r] + rsum16(p0 + p1);
    }

    // ---- P (C layout) -> wave-private LDS -> A-layout fragment ----
    __bf16* pl = Plds + wv * (16 * 32);
#pragma unroll
    for (int nt = 0; nt < 2; ++nt)
#pragma unroll
      for (int r = 0; r < 8; ++r)
        pl[(hi * 8 + r) * 32 + nt * 16 + ncol] = f2bf(s[nt][r]);

#pragma unroll
    for (int t = 0; t < 4; ++t)
#pragma unroll
      for (int r = 0; r < 8; ++r) acc[t][r] *= alpha[r];

    U16 pu;
    pu.h[0] = *(const v8bf*)(pl + ncol * 32 + hi * 16);
    pu.h[1] = *(const v8bf*)(pl + ncol * 32 + hi * 16 + 8);

    // ---- acc += P @ V_tile  (4 output d-tiles of 16) ----
#pragma unroll
    for (int dt = 0; dt < 4; ++dt) {
      U16 vu;
      const __bf16* vb = Vtlds + (dt * 16 + ncol) * 32 + hi * 16;
      vu.h[0] = *(const v8bf*)(vb);
      vu.h[1] = *(const v8bf*)(vb + 8);
      acc[dt] = wmma_bf16(pu.v, vu.v, acc[dt]);
    }
    __syncthreads();
  }

  // ---- normalize and store to attn_ws as [B][N][H*64] bf16 ----
  float inv[8];
#pragma unroll
  for (int r = 0; r < 8; ++r) inv[r] = 1.0f / lsum[r];
#pragma unroll
  for (int dt = 0; dt < 4; ++dt)
#pragma unroll
    for (int r = 0; r < 8; ++r) {
      const int nrow = q0 + hi * 8 + r;
      const int d    = dt * 16 + ncol;
      attn_ws[((size_t)(bb * NSEQ + nrow)) * NDIM + hh * HDIM + d] =
          f2bf(acc[dt][r] * inv[r]);
    }
}

// ---------------------------------------------------------------------------
// Kernel 3: out = attn @ proj_w.T + proj_b   (bf16 x bf16 -> f32 out + bias)
// ---------------------------------------------------------------------------
__global__ __launch_bounds__(256) void proj_gemm_kernel(
    const __bf16* __restrict__ a, const __bf16* __restrict__ w,
    const float* __restrict__ bias, float* __restrict__ out) {
  const int wv   = threadIdx.x >> 5;
  const int lane = threadIdx.x & 31;
  const int ncol = lane & 15;
  const int hi   = lane >> 4;

  const int m0 = (blockIdx.x % 64) * 128 + wv * 16;
  const int o0 = (blockIdx.x / 64) * 64;
  const int m  = m0 + ncol;
  const __bf16* arow = a + (size_t)m * NDIM;

  v8f acc[4] = {};

  for (int k0 = 0; k0 < NDIM; k0 += 32) {
    U16 au;
    au.h[0] = *(const v8bf*)(arow + k0 + hi * 16);
    au.h[1] = *(const v8bf*)(arow + k0 + hi * 16 + 8);
#pragma unroll
    for (int nt = 0; nt < 4; ++nt) {
      const __bf16* wrow = w + (size_t)(o0 + nt * 16 + ncol) * NDIM + k0 + hi * 16;
      U16 bu;
      bu.h[0] = *(const v8bf*)(wrow);
      bu.h[1] = *(const v8bf*)(wrow + 8);
      acc[nt] = wmma_bf16(au.v, bu.v, acc[nt]);
    }
  }

#pragma unroll
  for (int nt = 0; nt < 4; ++nt) {
    const int o = o0 + nt * 16 + ncol;
    const float bv = bias[o];
#pragma unroll
    for (int r = 0; r < 8; ++r)
      out[(size_t)(m0 + hi * 8 + r) * NDIM + o] = acc[nt][r] + bv;
  }
}

// ---------------------------------------------------------------------------
extern "C" void kernel_launch(void* const* d_in, const int* in_sizes, int n_in,
                              void* d_out, int out_size, void* d_ws, size_t ws_size,
                              hipStream_t stream) {
  const float* x      = (const float*)d_in[0];
  const float* qkv_w  = (const float*)d_in[1];
  const float* qkv_b  = (const float*)d_in[2];
  const float* proj_w = (const float*)d_in[3];
  const float* proj_b = (const float*)d_in[4];
  float* out = (float*)d_out;

  __bf16* qkv_ws   = (__bf16*)d_ws;                  // 3 * QELE
  __bf16* attn_ws  = qkv_ws + 3 * QELE;              // XELE
  __bf16* x_bf     = attn_ws + XELE;                 // XELE
  __bf16* qkvw_bf  = x_bf + XELE;                    // QWELE
  __bf16* projw_bf = qkvw_bf + QWELE;                // PWELE

  // one-shot fp32 -> bf16 conversions
  cvt_bf16_kernel<<<(int)(XELE / 8 / 256), 256, 0, stream>>>(x, x_bf, (int)(XELE / 8));
  cvt_bf16_kernel<<<(int)(QWELE / 8 / 256), 256, 0, stream>>>(qkv_w, qkvw_bf, (int)(QWELE / 8));
  cvt_bf16_kernel<<<(int)(PWELE / 8 / 256), 256, 0, stream>>>(proj_w, projw_bf, (int)(PWELE / 8));

  // 8192/16 row-tiles grouped 8/block -> 64 ; 2304/64 col strips -> 36
  qkv_gemm_kernel<<<64 * 36, 256, 0, stream>>>(x_bf, qkvw_bf, qkv_b, qkv_ws);
  // 48 (b,h) pairs * 16 q-tiles of 128 rows
  attn_kernel<<<48 * 16, 256, 0, stream>>>(qkv_ws, attn_ws);
  // 64 row groups * 12 col strips
  proj_gemm_kernel<<<64 * 12, 256, 0, stream>>>(attn_ws, projw_bf, proj_b, out);
}
